// NonLocalBlock_41549513622357
// MI455X (gfx1250) — matrix operand order repaired and checked by reading, not compiled, not run on hardware
//
#include <hip/hip_runtime.h>

#define EPS  1e-5f
#define CIN  256
#define FDIM 128
#define NPIX 4096

typedef __attribute__((ext_vector_type(16))) _Float16 v16h;
typedef __attribute__((ext_vector_type(8)))  _Float16 v8h;
typedef __attribute__((ext_vector_type(8)))  float    v8f;
typedef __attribute__((ext_vector_type(4)))  int      v4i;

#define AS1 __attribute__((address_space(1)))
#define AS3 __attribute__((address_space(3)))

// Build a 16-half A/B fragment from two contiguous 16-byte chunks (global or LDS).
__device__ __forceinline__ v16h pack16(const _Float16* lo, const _Float16* hi) {
  v8h a = *reinterpret_cast<const v8h*>(lo);
  v8h b = *reinterpret_cast<const v8h*>(hi);
  v16h r;
#pragma unroll
  for (int i = 0; i < 8; ++i) { r[i] = a[i]; r[i + 8] = b[i]; }
  return r;
}

// 16-byte global -> LDS copy at compile-time byte offset OFF (added to both sides,
// matching the ISA's IOFFSET semantics). Async DMA if the toolchain exposes it.
template <int OFF>
__device__ __forceinline__ void cp16(const _Float16* g, _Float16* l) {
#if __has_builtin(__builtin_amdgcn_global_load_async_to_lds_b128)
  __builtin_amdgcn_global_load_async_to_lds_b128(
      (AS1 v4i*)g, (AS3 v4i*)l, OFF, 0);
#else
  *reinterpret_cast<v8h*>(reinterpret_cast<char*>(l) + OFF) =
      *reinterpret_cast<const v8h*>(reinterpret_cast<const char*>(g) + OFF);
#endif
}

__device__ __forceinline__ void wait_async() {
#if __has_builtin(__builtin_amdgcn_s_wait_asynccnt)
  __builtin_amdgcn_s_wait_asynccnt(0);
#endif
}

// ---------------------------------------------------------------------------
// Kernel 1: 1x1 conv + BN(eval) + ReLU -> f16, optionally transposed layout.
//   transposed=1: dst[b][n][f]  (for t / p, f contiguous -> b128 frag loads)
//   transposed=0: dst[b][f][n]  (for z, m contiguous -> b128 frag loads)
// ---------------------------------------------------------------------------
__global__ __launch_bounds__(256) void proj_cna(
    const float* __restrict__ x, const float* __restrict__ W,
    const float* __restrict__ bconv, const float* __restrict__ gam,
    const float* __restrict__ bet, const float* __restrict__ mu,
    const float* __restrict__ var, _Float16* __restrict__ dst, int transposed) {
  __shared__ float sx[CIN * 32];
  const int b  = blockIdx.y;
  const int n0 = blockIdx.x * 32;
  const int t  = threadIdx.x;
#pragma unroll
  for (int r = 0; r < 32; ++r) {
    int idx = t + r * 256;
    int c = idx >> 5, j = idx & 31;
    sx[idx] = x[((size_t)(b * CIN + c)) * NPIX + n0 + j];
  }
  __syncthreads();
  const int f  = t & 127;
  const int jh = (t >> 7) * 16;
  float acc[16];
#pragma unroll
  for (int k = 0; k < 16; ++k) acc[k] = 0.f;
  for (int c = 0; c < CIN; ++c) {
    float w = W[f * CIN + c];
    const float* sp = &sx[c * 32 + jh];
#pragma unroll
    for (int k = 0; k < 16; ++k) acc[k] = fmaf(w, sp[k], acc[k]);
  }
  float scale = gam[f] * rsqrtf(var[f] + EPS);
  float shift = (bconv[f] - mu[f]) * scale + bet[f];
#pragma unroll
  for (int k = 0; k < 16; ++k) {
    float v = fmaf(acc[k], scale, shift);
    v = v > 0.f ? v : 0.f;
    int n = n0 + jh + k;
    if (transposed)
      dst[((size_t)(b * NPIX + n)) * FDIM + f] = (_Float16)v;
    else
      dst[((size_t)(b * FDIM + f)) * NPIX + n] = (_Float16)v;
  }
}

// ---------------------------------------------------------------------------
// Kernel 2: fused flash attention. One wave = 16 query positions; the 4 waves
// of a block share async-DMA-staged, double-buffered p/z tiles in LDS.
// ---------------------------------------------------------------------------
__global__ __launch_bounds__(128) void flash_attn(
    const _Float16* __restrict__ tT, const _Float16* __restrict__ pT,
    const _Float16* __restrict__ zb, float* __restrict__ u) {
  __shared__ _Float16 pS[2][32 * FDIM];   // 8 KB per buffer: p[m_rel][f]
  __shared__ _Float16 zS[2][FDIM * 32];   // 8 KB per buffer: z[f][m_rel]
  const int t    = threadIdx.x;
  const int wave = t >> 5;
  const int lane = t & 31;
  const int llo  = lane & 15;
  const int lhi  = lane >> 4;
  const int b    = blockIdx.y;
  const int nb   = (blockIdx.x * 4 + wave) * 16;

  // t as B operand (K=f axis): lane holds n=llo, halfs h -> f = s*32 + (h&7) + 16*(h>>3) + 8*lhi
  v16h tB[4];
  {
    const _Float16* tr = tT + ((size_t)(b * NPIX + nb + llo)) * FDIM + lhi * 8;
#pragma unroll
    for (int s = 0; s < 4; ++s) tB[s] = pack16(tr + s * 32, tr + s * 32 + 16);
  }

  // Stage the 32-m-wide p and z tiles (shared by all 4 waves) into LDS.
  // p tile is one contiguous 8KB block (pT is [n][f]); z tile is 128 rows of 64B.
  auto prefetch = [&](int buf, int mb) {
    const _Float16* pg = pT + ((size_t)(b * NPIX + mb)) * FDIM + t * 8;
    _Float16* pl = &pS[buf][t * 8];
    cp16<0>(pg, pl); cp16<2048>(pg, pl); cp16<4096>(pg, pl); cp16<6144>(pg, pl);
    const _Float16* zg = zb + ((size_t)(b * FDIM + t)) * NPIX + mb;
    _Float16* zl = &zS[buf][t * 32];
    cp16<0>(zg, zl); cp16<16>(zg, zl); cp16<32>(zg, zl); cp16<48>(zg, zl);
  };

  v8f zero = {};
  v8f O[8];
#pragma unroll
  for (int fb = 0; fb < 8; ++fb) O[fb] = zero;
  float rmax = -1e30f, rsum = 0.f;

  prefetch(0, 0);
  for (int mb = 0; mb < NPIX; mb += 32) {
    const int cur = (mb >> 5) & 1;
    wait_async();        // own DMAs for buffer `cur` complete
    __syncthreads();     // everyone's DMAs landed; prior reads of other buffer done
    if (mb + 32 < NPIX) prefetch(cur ^ 1, mb + 32);

    // p as A operand from LDS: lane row m = mb(+16)+llo; halfs f = s*32+(h&7)+16*(h>>3)+8*lhi
    const _Float16* p0 = &pS[cur][llo * FDIM + lhi * 8];
    const _Float16* p1 = &pS[cur][(16 + llo) * FDIM + lhi * 8];
    v8f S0 = zero, S1 = zero;
#pragma unroll
    for (int s = 0; s < 4; ++s) {
      v16h a0 = pack16(p0 + s * 32, p0 + s * 32 + 16);
      v16h a1 = pack16(p1 + s * 32, p1 + s * 32 + 16);
      S0 = __builtin_amdgcn_wmma_f32_16x16x32_f16(false, a0, false, tB[s], (short)0, S0, false, false);
      S1 = __builtin_amdgcn_wmma_f32_16x16x32_f16(false, a1, false, tB[s], (short)0, S1, false, false);
    }
    // S-tile: lane holds n=llo, m = mb + v + 8*lhi (S0) / +16 (S1)
    float tmax = S0[0];
#pragma unroll
    for (int v = 1; v < 8; ++v) tmax = fmaxf(tmax, S0[v]);
#pragma unroll
    for (int v = 0; v < 8; ++v) tmax = fmaxf(tmax, S1[v]);
    tmax = fmaxf(tmax, __shfl_xor(tmax, 16, 32));
    float nmax  = fmaxf(rmax, tmax);
    float alpha = __expf(rmax - nmax);
    rmax = nmax;
    rsum *= alpha;
#pragma unroll
    for (int fb = 0; fb < 8; ++fb) O[fb] *= alpha;

    float P0[8], P1[8], ls = 0.f;
#pragma unroll
    for (int v = 0; v < 8; ++v) { P0[v] = __expf(S0[v] - rmax); ls += P0[v]; }
#pragma unroll
    for (int v = 0; v < 8; ++v) { P1[v] = __expf(S1[v] - rmax); ls += P1[v]; }
    rsum += ls + __shfl_xor(ls, 16, 32);

    // D-layout -> B-operand layout (K=m 32, N=n 16) via half-lane swap + f16 pack.
    v16h B2;
#pragma unroll
    for (int v = 0; v < 8; ++v) {
      float o0 = __shfl_xor(P0[v], 16, 32);
      float o1 = __shfl_xor(P1[v], 16, 32);
      B2[v]     = (_Float16)(lhi ? o1 : P0[v]);
      B2[v + 8] = (_Float16)(lhi ? P1[v] : o0);
    }

    // z as A operand from LDS: lane row f = fb*16+llo; halfs m = h+16*lhi
#pragma unroll
    for (int fb = 0; fb < 8; ++fb) {
      const _Float16* zr = &zS[cur][(fb * 16 + llo) * 32 + lhi * 16];
      v16h zA = pack16(zr, zr + 8);
      O[fb] = __builtin_amdgcn_wmma_f32_16x16x32_f16(false, zA, false, B2, (short)0, O[fb], false, false);
    }
  }

  // normalize and store u[b][f][n]  (O: lane n=llo, f = fb*16 + v + 8*lhi)
  float inv = 1.f / rsum;
#pragma unroll
  for (int fb = 0; fb < 8; ++fb) {
    int f = fb * 16 + 8 * lhi;
#pragma unroll
    for (int v = 0; v < 8; ++v)
      u[((size_t)(b * FDIM + f + v)) * NPIX + nb + llo] = O[fb][v] * inv;
  }
}

// ---------------------------------------------------------------------------
// Kernel 3: pre = pre_w @ u + pre_b; out = relu(bn(pre + x))
// ---------------------------------------------------------------------------
__global__ __launch_bounds__(256) void out_proj(
    const float* __restrict__ u, const float* __restrict__ x,
    const float* __restrict__ pw, const float* __restrict__ pb,
    const float* __restrict__ gam, const float* __restrict__ bet,
    const float* __restrict__ mu, const float* __restrict__ var,
    float* __restrict__ out) {
  __shared__ float su[FDIM * 32];
  const int b  = blockIdx.y;
  const int n0 = blockIdx.x * 32;
  const int t  = threadIdx.x;
#pragma unroll
  for (int r = 0; r < 16; ++r) {
    int idx = t + r * 256;
    int f = idx >> 5, j = idx & 31;
    su[idx] = u[((size_t)(b * FDIM + f)) * NPIX + n0 + j];
  }
  __syncthreads();
  const int c = t;
  float acc[32];
#pragma unroll
  for (int j = 0; j < 32; ++j) acc[j] = 0.f;
  for (int f = 0; f < FDIM; ++f) {
    float w = pw[c * FDIM + f];
    const float* sp = &su[f * 32];
#pragma unroll
    for (int j = 0; j < 32; ++j) acc[j] = fmaf(w, sp[j], acc[j]);
  }
  float scale = gam[c] * rsqrtf(var[c] + EPS);
  float shift = bet[c] - mu[c] * scale;
  float bias  = pb[c];
  const size_t base = ((size_t)(b * CIN + c)) * NPIX + n0;
#pragma unroll
  for (int j = 0; j < 32; ++j) {
    float v = fmaf(acc[j] + bias + x[base + j], scale, shift);
    out[base + j] = v > 0.f ? v : 0.f;
  }
}

extern "C" void kernel_launch(void* const* d_in, const int* in_sizes, int n_in,
                              void* d_out, int out_size, void* d_ws, size_t ws_size,
                              hipStream_t stream) {
  (void)in_sizes; (void)n_in; (void)out_size; (void)ws_size;
  const float* x     = (const float*)d_in[0];
  const float* th_w  = (const float*)d_in[1];  const float* th_b  = (const float*)d_in[2];
  const float* th_g  = (const float*)d_in[3];  const float* th_be = (const float*)d_in[4];
  const float* th_m  = (const float*)d_in[5];  const float* th_v  = (const float*)d_in[6];
  const float* ph_w  = (const float*)d_in[7];  const float* ph_b  = (const float*)d_in[8];
  const float* ph_g  = (const float*)d_in[9];  const float* ph_be = (const float*)d_in[10];
  const float* ph_m  = (const float*)d_in[11]; const float* ph_v  = (const float*)d_in[12];
  const float* mu_w  = (const float*)d_in[13]; const float* mu_b  = (const float*)d_in[14];
  const float* mu_g  = (const float*)d_in[15]; const float* mu_be = (const float*)d_in[16];
  const float* mu_m  = (const float*)d_in[17]; const float* mu_v  = (const float*)d_in[18];
  const float* pre_w = (const float*)d_in[19]; const float* pre_b = (const float*)d_in[20];
  const float* po_g  = (const float*)d_in[21]; const float* po_be = (const float*)d_in[22];
  const float* po_m  = (const float*)d_in[23]; const float* po_v  = (const float*)d_in[24];

  const size_t SZH = (size_t)8 * NPIX * FDIM;
  _Float16* tT = (_Float16*)d_ws;
  _Float16* pT = tT + SZH;
  _Float16* zb = pT + SZH;
  float*    u  = (float*)(zb + SZH);

  dim3 gProj(NPIX / 32, 8);
  proj_cna<<<gProj, 256, 0, stream>>>(x, th_w, th_b, th_g, th_be, th_m, th_v, tT, 1);
  proj_cna<<<gProj, 256, 0, stream>>>(x, ph_w, ph_b, ph_g, ph_be, ph_m, ph_v, pT, 1);
  proj_cna<<<gProj, 256, 0, stream>>>(x, mu_w, mu_b, mu_g, mu_be, mu_m, mu_v, zb, 0);

  flash_attn<<<dim3(NPIX / 64, 8), 128, 0, stream>>>(tT, pT, zb, u);

  out_proj<<<dim3(NPIX / 32, 8), 256, 0, stream>>>(u, x, pre_w, pre_b, po_g, po_be, po_m, po_v,
                                                   (float*)d_out);
}
